// CTSelfAttention_7146825580839
// MI455X (gfx1250) — matrix operand order repaired and checked by reading, not compiled, not run on hardware
//
#include <hip/hip_runtime.h>
#include <hip/hip_bf16.h>
#include <stdint.h>

// ---------------------------------------------------------------------------
// CT self-attention for MI455X (gfx1250, wave32, WMMA + Tensor Data Mover).
//   B=4, T=2048, D=1024, H=16, HD=64, L=2 (causal + 2 future frames)
// Pipeline:
//   0) convert x / w_qkv / w_out fp32 -> f16 (one pass; halves GEMM HBM bytes)
//   1) qkv = x @ w_qkv^T + b_qkv    (TDM tile staging -> LDS, f16 WMMA)
//   2) flash-style masked attention (f16 WMMA, online softmax)
//   3) out = attn @ w_out^T + b_out (TDM staging for W, f16 WMMA, fp32 out)
// ---------------------------------------------------------------------------

typedef __attribute__((ext_vector_type(16))) _Float16     v16h;
typedef __attribute__((ext_vector_type(8)))  _Float16     v8h;
typedef __attribute__((ext_vector_type(8)))  float        v8f;
typedef __attribute__((ext_vector_type(4)))  unsigned int u32x4;
typedef __attribute__((ext_vector_type(8)))  int          i32x8;
typedef __attribute__((ext_vector_type(4)))  int          i32x4;

static __device__ __forceinline__ v8f wmma_f16(v16h a, v16h b, v8f c) {
  // D = A(16x32,f16) * B(32x16,f16) + C(16x16,f32)
  return __builtin_amdgcn_wmma_f32_16x16x32_f16(
      /*neg_a=*/false, a, /*neg_b=*/false, b,
      /*c_mod=*/(short)0, c, /*reuse_a=*/false, /*reuse_b=*/false);
}

// Tensor Data Mover: 2D f16 tile (tile_d1 rows x tile_d0 elems) -> LDS.
// D# built per CDNA5 ISA ch.8: group0 {count, lds_addr, global_addr, type=2},
// group1 {data_size=2B, tensor dims, tile dims, dim0 stride}; groups 2/3 = 0.
static __device__ __forceinline__ void tdm_load_2d_f16(
    unsigned lds_off, const void* gptr, unsigned tensor_d0, unsigned tensor_d1,
    unsigned tile_d0, unsigned tile_d1, unsigned stride0) {
  const unsigned long long ga = (unsigned long long)gptr;
  u32x4 g0;
  g0[0] = 1u;                                         // count=1, user mode
  g0[1] = lds_off;                                    // LDS byte address
  g0[2] = (unsigned)(ga & 0xFFFFFFFFull);             // global_addr[31:0]
  g0[3] = (unsigned)((ga >> 32) & 0x01FFFFFFull)      // global_addr[56:32]
          | (2u << 30);                               // type=2 ("image")
  i32x8 g1;
  g1[0] = (int)(1u << 16);                            // data_size=1 -> 2 bytes
  g1[1] = (int)((tensor_d0 & 0xFFFFu) << 16);         // tensor_dim0[15:0]
  g1[2] = (int)((tensor_d0 >> 16) | ((tensor_d1 & 0xFFFFu) << 16));
  g1[3] = (int)((tensor_d1 >> 16) | (tile_d0 << 16)); // tile_dim0
  g1[4] = (int)(tile_d1 & 0xFFFFu);                   // tile_dim1, tile_dim2=0
  g1[5] = (int)stride0;                               // tensor_dim0_stride lo
  g1[6] = 0;
  g1[7] = 0;
  i32x4 z4 = {0, 0, 0, 0};
#if defined(__clang_major__) && (__clang_major__ >= 23)
  i32x8 z8 = {0, 0, 0, 0, 0, 0, 0, 0};
  __builtin_amdgcn_tensor_load_to_lds(g0, g1, z4, z4, z8, 0);
#else
  __builtin_amdgcn_tensor_load_to_lds(g0, g1, z4, z4, 0);
#endif
}

// Problem constants
constexpr int CB  = 4;
constexpr int CT  = 2048;
constexpr int CD  = 1024;
constexpr int CH  = 16;
constexpr int CHD = 64;
constexpr int CL  = 2;

constexpr int M1 = CB * CT;   // 8192 rows of x
constexpr int N1 = 3 * CD;    // 3072 qkv outputs
constexpr int K1 = CD;        // 1024

// GEMM tiling: block = 256 threads = 8 waves; tile 128(M) x 64(N), K-step 32.
constexpr int TM = 128, TN = 64, TK = 32;

// ---------------------------------------------------------------------------
// Kernel 0: elementwise fp32 -> f16 (n must be a multiple of 8)
// ---------------------------------------------------------------------------
__global__ __launch_bounds__(256) void cvt_f32_f16(
    const float* __restrict__ in, _Float16* __restrict__ out, int n) {
  const int i = (blockIdx.x * 256 + threadIdx.x) * 8;
  if (i < n) {
    const float4* p = (const float4*)(in + i);
    const float4 a = p[0], b = p[1];
    v8h h;
    h[0] = (_Float16)a.x; h[1] = (_Float16)a.y;
    h[2] = (_Float16)a.z; h[3] = (_Float16)a.w;
    h[4] = (_Float16)b.x; h[5] = (_Float16)b.y;
    h[6] = (_Float16)b.z; h[7] = (_Float16)b.w;
    *(v8h*)(out + i) = h;
  }
}

// ---------------------------------------------------------------------------
// Kernel 1: qkv = x @ w_qkv^T + b_qkv   -> f16 [M1, N1] (A,B staged by TDM)
// ---------------------------------------------------------------------------
__global__ __launch_bounds__(256) void qkv_gemm_f16(
    const _Float16* __restrict__ xh, const _Float16* __restrict__ wh,
    const float* __restrict__ bias, _Float16* __restrict__ out) {
  __shared__ _Float16 lA[TM][TK];  // 8 KB
  __shared__ _Float16 lB[TN][TK];  // 4 KB

  const int tid  = threadIdx.x;
  const int lane = tid & 31;
  const int wid  = tid >> 5;
  const int m0   = blockIdx.x * TM;
  const int n0   = blockIdx.y * TN;
  const int wm   = (wid >> 1) * 32;  // wave M offset in tile: 0..96
  const int wn   = (wid & 1) * 32;   // wave N offset in tile: 0,32
  const int fr   = lane & 15;
  const int fh   = (lane >> 4) * 16;

  const unsigned ldsA = (unsigned)(uintptr_t)&lA[0][0];
  const unsigned ldsB = (unsigned)(uintptr_t)&lB[0][0];

  v8f acc[2][2] = {};

  for (int k0 = 0; k0 < K1; k0 += TK) {
    if (wid == 0) {
      // DMA both tiles into LDS; wave0 drains TENSORcnt, barrier publishes.
      tdm_load_2d_f16(ldsA, &xh[(size_t)m0 * K1 + k0], K1, M1, TK, TM, K1);
      tdm_load_2d_f16(ldsB, &wh[(size_t)n0 * K1 + k0], K1, N1, TK, TN, K1);
      __builtin_amdgcn_s_wait_tensorcnt(0);
    }
    __syncthreads();

    v16h a0 = *(const v16h*)&lA[wm + fr][fh];
    v16h a1 = *(const v16h*)&lA[wm + 16 + fr][fh];
    v16h b0 = *(const v16h*)&lB[wn + fr][fh];
    v16h b1 = *(const v16h*)&lB[wn + 16 + fr][fh];

    acc[0][0] = wmma_f16(a0, b0, acc[0][0]);
    acc[0][1] = wmma_f16(a0, b1, acc[0][1]);
    acc[1][0] = wmma_f16(a1, b0, acc[1][0]);
    acc[1][1] = wmma_f16(a1, b1, acc[1][1]);
    __syncthreads();  // protect LDS against next TDM write (WAR)
  }

  // Epilogue: C layout -> rows (vgpr e, +8 for lanes 16..31), cols (lane&15)
#pragma unroll
  for (int i = 0; i < 2; ++i) {
#pragma unroll
    for (int j = 0; j < 2; ++j) {
      const int nb = n0 + wn + j * 16 + fr;
      const float bv = bias[nb];
      const int rb = m0 + wm + i * 16 + (lane >> 4) * 8;
#pragma unroll
      for (int e = 0; e < 8; ++e)
        out[(size_t)(rb + e) * N1 + nb] = (_Float16)(acc[i][j][e] + bv);
    }
  }
}

// ---------------------------------------------------------------------------
// Kernel 2: flash-style masked attention.
//   qkv: f16 [B,T, 3*D] with n = s*D + h*HD + hd ; attn: f16 [B,T,D]
// One wave handles 16 query rows; key loop in steps of 32.
// ---------------------------------------------------------------------------
__global__ __launch_bounds__(256) void ct_attention(
    const _Float16* __restrict__ qkv, const unsigned char* __restrict__ kpm,
    _Float16* __restrict__ attn) {
  __shared__ _Float16 lP[8][16][32];  // per-wave P staging (8 KB total)

  const int lane  = threadIdx.x & 31;
  const int wid   = threadIdx.x >> 5;
  const int b     = blockIdx.z;
  const int h     = blockIdx.y;
  const int qtile = blockIdx.x * 8 + wid;  // 0..127
  const int qbase = qtile * 16;
  const int fr    = lane & 15;
  const int fh    = (lane >> 4) * 16;
  const float scale = 0.125f;  // 1/sqrt(64)
  const size_t rowS = (size_t)(3 * CD);
  const size_t base = (size_t)b * CT;

  // Q fragments: 16 rows x 64 (HD) -> two 16x32 f16 A-fragments
  const v16h aq0 = *(const v16h*)&qkv[(base + qbase + fr) * rowS + h * CHD + fh];
  const v16h aq1 = *(const v16h*)&qkv[(base + qbase + fr) * rowS + h * CHD + 32 + fh];

  v8f o[4] = {};
  float mi[8], li[8];
#pragma unroll
  for (int e = 0; e < 8; ++e) { mi[e] = -1e30f; li[e] = 0.f; }

  int nkeys = qbase + 16 + CL;  // last row qbase+15 sees keys <= row+L
  if (nkeys > CT) nkeys = CT;
  const int ntiles = (nkeys + 31) / 32;
  const int rowg = qbase + (lane >> 4) * 8;  // + e = this lane's row ids

  for (int kt = 0; kt < ntiles; ++kt) {
    const int kb = kt * 32;
    if (kt + 1 < ntiles) {  // prefetch next K tile (global_prefetch_b8)
      __builtin_prefetch(&qkv[(base + kb + 32 + lane) * rowS + CD + h * CHD],
                         0, 1);
    }

    // ---- scores: S = Q K^T for 32 keys (two 16-key column tiles) ----
    v8f s0 = {}, s1 = {};
    {
      const size_t kr0 = (base + kb + fr) * rowS + CD + h * CHD;
      v16h bk0 = *(const v16h*)&qkv[kr0 + fh];
      v16h bk1 = *(const v16h*)&qkv[kr0 + 32 + fh];
      s0 = wmma_f16(aq0, bk0, s0);
      s0 = wmma_f16(aq1, bk1, s0);
      const size_t kr1 = (base + kb + 16 + fr) * rowS + CD + h * CHD;
      bk0 = *(const v16h*)&qkv[kr1 + fh];
      bk1 = *(const v16h*)&qkv[kr1 + 32 + fh];
      s1 = wmma_f16(aq0, bk0, s1);
      s1 = wmma_f16(aq1, bk1, s1);
    }

    // ---- mask + scale + online softmax ----
    const int c0 = kb + fr;
    const int c1 = kb + 16 + fr;
    const bool pad0 = kpm[b * CT + c0] != 0;
    const bool pad1 = kpm[b * CT + c1] != 0;

    float p0[8], p1[8], mnew[8];
#pragma unroll
    for (int e = 0; e < 8; ++e) {
      float v0 = s0[e] * scale;
      float v1 = s1[e] * scale;
      if (c0 > rowg + e + CL || pad0) v0 = -1e30f;
      if (c1 > rowg + e + CL || pad1) v1 = -1e30f;
      p0[e] = v0; p1[e] = v1;
      mnew[e] = fmaxf(v0, v1);
    }
    // row-max over the 16 lanes sharing a row (xor masks stay in-half)
#pragma unroll
    for (int d = 1; d < 16; d <<= 1)
#pragma unroll
      for (int e = 0; e < 8; ++e)
        mnew[e] = fmaxf(mnew[e], __shfl_xor(mnew[e], d));

    float corr[8], rs[8];
#pragma unroll
    for (int e = 0; e < 8; ++e) {
      mnew[e] = fmaxf(mnew[e], mi[e]);
      corr[e] = __expf(mi[e] - mnew[e]);
      mi[e]   = mnew[e];
      p0[e]   = __expf(p0[e] - mnew[e]);
      p1[e]   = __expf(p1[e] - mnew[e]);
      rs[e]   = p0[e] + p1[e];
    }
#pragma unroll
    for (int d = 1; d < 16; d <<= 1)
#pragma unroll
      for (int e = 0; e < 8; ++e) rs[e] += __shfl_xor(rs[e], d);
#pragma unroll
    for (int e = 0; e < 8; ++e) li[e] = li[e] * corr[e] + rs[e];
#pragma unroll
    for (int n = 0; n < 4; ++n)
#pragma unroll
      for (int e = 0; e < 8; ++e) o[n][e] *= corr[e];

    // ---- reshape P (C-layout) -> A-fragment via per-wave LDS patch ----
    const int rbase = (lane >> 4) * 8;
#pragma unroll
    for (int e = 0; e < 8; ++e) {
      lP[wid][rbase + e][fr]      = (_Float16)p0[e];
      lP[wid][rbase + e][16 + fr] = (_Float16)p1[e];
    }
    const v16h pa = *(const v16h*)&lP[wid][fr][fh];

    // ---- O += P * V over four 16-wide HD chunks ----
#pragma unroll
    for (int n = 0; n < 4; ++n) {
      v16h bv;
#pragma unroll
      for (int s = 0; s < 16; ++s)
        bv[s] = qkv[(base + kb + fh + s) * rowS + 2 * CD + h * CHD + n * 16 + fr];
      o[n] = wmma_f16(pa, bv, o[n]);
    }
  }

  // ---- finalize: divide by row sum, store f16 [B,T,D] ----
#pragma unroll
  for (int n = 0; n < 4; ++n) {
#pragma unroll
    for (int e = 0; e < 8; ++e) {
      const int t = rowg + e;
      const float val = o[n][e] / li[e];
      attn[(base + t) * CD + h * CHD + n * 16 + fr] = (_Float16)val;
    }
  }
}

// ---------------------------------------------------------------------------
// Kernel 3: out = attn @ w_out^T + b_out
//   A (attn) is f16: direct global fragments. B (w_out f16) staged via TDM.
// ---------------------------------------------------------------------------
__global__ __launch_bounds__(256) void out_gemm(
    const _Float16* __restrict__ a, const _Float16* __restrict__ wh,
    const float* __restrict__ bias, float* __restrict__ out) {
  __shared__ _Float16 lB[TN][TK];

  const int tid  = threadIdx.x;
  const int lane = tid & 31;
  const int wid  = tid >> 5;
  const int m0   = blockIdx.x * TM;
  const int n0   = blockIdx.y * TN;
  const int wm   = (wid >> 1) * 32;
  const int wn   = (wid & 1) * 32;
  const int fr   = lane & 15;
  const int fh   = (lane >> 4) * 16;

  const unsigned ldsB = (unsigned)(uintptr_t)&lB[0][0];

  v8f acc[2][2] = {};

  for (int k0 = 0; k0 < CD; k0 += TK) {
    if (wid == 0) {
      tdm_load_2d_f16(ldsB, &wh[(size_t)n0 * CD + k0], CD, CD, TK, TN, CD);
      __builtin_amdgcn_s_wait_tensorcnt(0);
    }
    __syncthreads();

    // A fragments straight from global f16 (32B-aligned b128 pairs)
    v16h a0 = *(const v16h*)&a[(size_t)(m0 + wm + fr) * CD + k0 + fh];
    v16h a1 = *(const v16h*)&a[(size_t)(m0 + wm + 16 + fr) * CD + k0 + fh];
    v16h b0 = *(const v16h*)&lB[wn + fr][fh];
    v16h b1 = *(const v16h*)&lB[wn + 16 + fr][fh];

    acc[0][0] = wmma_f16(a0, b0, acc[0][0]);
    acc[0][1] = wmma_f16(a0, b1, acc[0][1]);
    acc[1][0] = wmma_f16(a1, b0, acc[1][0]);
    acc[1][1] = wmma_f16(a1, b1, acc[1][1]);
    __syncthreads();
  }

#pragma unroll
  for (int i = 0; i < 2; ++i) {
#pragma unroll
    for (int j = 0; j < 2; ++j) {
      const int nb = n0 + wn + j * 16 + fr;
      const float bv = bias[nb];
      const int rb = m0 + wm + i * 16 + (lane >> 4) * 8;
#pragma unroll
      for (int e = 0; e < 8; ++e)
        out[(size_t)(rb + e) * CD + nb] = acc[i][j][e] + bv;
    }
  }
}

// ---------------------------------------------------------------------------
extern "C" void kernel_launch(void* const* d_in, const int* in_sizes, int n_in,
                              void* d_out, int out_size, void* d_ws,
                              size_t ws_size, hipStream_t stream) {
  const float* x           = (const float*)d_in[0];
  const unsigned char* kpm = (const unsigned char*)d_in[1];  // bool [B,T]
  const float* w_qkv       = (const float*)d_in[2];
  const float* b_qkv       = (const float*)d_in[3];
  const float* w_out       = (const float*)d_in[4];
  const float* b_out       = (const float*)d_in[5];
  float* out               = (float*)d_out;

  // Workspace layout (f16): qkv 48MB | attn 16MB | x 16MB | w_qkv 6MB | w_out 2MB
  char* ws = (char*)d_ws;
  _Float16* qkv_f16  = (_Float16*)ws;
  _Float16* attn_f16 = (_Float16*)(ws + (size_t)M1 * N1 * 2);
  _Float16* x_f16    = (_Float16*)(ws + (size_t)M1 * N1 * 2 + (size_t)M1 * CD * 2);
  _Float16* wqkv_f16 = (_Float16*)((char*)x_f16 + (size_t)M1 * K1 * 2);
  _Float16* wout_f16 = (_Float16*)((char*)wqkv_f16 + (size_t)N1 * K1 * 2);

  // 0) one-shot fp32 -> f16 conversions
  cvt_f32_f16<<<(M1 * K1 / 8 + 255) / 256, 256, 0, stream>>>(x, x_f16, M1 * K1);
  cvt_f32_f16<<<(N1 * K1 / 8 + 255) / 256, 256, 0, stream>>>(w_qkv, wqkv_f16,
                                                             N1 * K1);
  cvt_f32_f16<<<(CD * CD / 8 + 255) / 256, 256, 0, stream>>>(w_out, wout_f16,
                                                             CD * CD);
  // 1) QKV projection (TDM-staged tiles)
  qkv_gemm_f16<<<dim3(M1 / TM, N1 / TN), 256, 0, stream>>>(x_f16, wqkv_f16,
                                                           b_qkv, qkv_f16);
  // 2) Masked attention (grid: 16 q-tile groups x H x B; 8 waves/block)
  ct_attention<<<dim3(CT / 16 / 8, CH, CB), 256, 0, stream>>>(qkv_f16, kpm,
                                                              attn_f16);
  // 3) Output projection (TDM-staged W)
  out_gemm<<<dim3(M1 / TM, CD / TN), 256, 0, stream>>>(attn_f16, wout_f16,
                                                       b_out, out);
}
